// MAS_GNFE_16999480557909
// MI455X (gfx1250) — matrix operand order repaired and checked by reading, not compiled, not run on hardware
//
#include <hip/hip_runtime.h>
#include <hip/hip_bf16.h>

// ---------------------------------------------------------------------------
// MAS+GNFE fused pipeline for MI455X (gfx1250, wave32, WMMA bf16 + TDM)
// ---------------------------------------------------------------------------

constexpr int N = 4096;
constexpr int D = 64;
constexpr int L3 = 3;

typedef __attribute__((ext_vector_type(16))) __bf16 bf16x16;
typedef __attribute__((ext_vector_type(8)))  float  v8f;
typedef __attribute__((ext_vector_type(4)))  unsigned int u32x4;
typedef __attribute__((ext_vector_type(8)))  int          i32x8;
typedef __attribute__((ext_vector_type(4)))  int          i32x4;

union FragBF {                 // one WMMA 16-bit A/B operand (8 VGPRs/lane)
    bf16x16 v;
    __bf16  e[16];
    unsigned int u[8];
};
union Pack8 {                  // 8 contiguous bf16 = one 128-bit store
    uint4  q;
    __bf16 e[8];
};

__device__ __forceinline__ float lrelu(float x) { return x > 0.f ? x : 0.2f * x; }

// A-matrix (MxK, 16-bit) lane layout (contiguous, matches B panels):
//   element i -> K = 16h + i
__device__ __forceinline__ int kmapA(int i, int h) {
    return 16 * h + i;
}

// LDS byte offset of a __shared__ object: LDS aperture maps addr[31:0] -> offset
__device__ __forceinline__ unsigned int lds_offset(const void* p) {
    return (unsigned int)(uintptr_t)p;
}

// Issue a TDM load of a 64(row=d) x 32(col=j) bf16 tile from a (D x N) bf16
// tensor starting at gbase (tile start), into LDS at ldsoff (row-major 64x32).
// D# built per ISA ch.8: data_size=2B, tensor_dim0=N, dim0_stride=N,
// tensor_dim1=D, tile_dim0=32, tile_dim1=64, 2D tile (groups 2/3 zero).
__device__ __forceinline__ void tdm_load_panel(const __bf16* gbase,
                                               unsigned int ldsoff) {
    unsigned long long ga = (unsigned long long)(uintptr_t)gbase;
    u32x4 g0;
    g0[0] = 1u;                                   // count=1, user mode, no gather
    g0[1] = ldsoff;                               // lds_addr [63:32]
    g0[2] = (unsigned int)ga;                     // global_addr[31:0]
    g0[3] = (unsigned int)(ga >> 32) | (2u << 30);// global_addr[56:32] | type=2
    i32x8 g1;
    g1[0] = 0x10000;                              // data_size=1 (2 bytes)
    g1[1] = (int)(4096u << 16);                   // tensor_dim0[15:0]=4096
    g1[2] = (int)(64u << 16);                     // tensor_dim0_hi=0 | tensor_dim1=64
    g1[3] = (int)(32u << 16);                     // tensor_dim1_hi=0 | tile_dim0=32
    g1[4] = 64;                                   // tile_dim1=64, tile_dim2=0
    g1[5] = 4096;                                 // tensor_dim0_stride lo
    g1[6] = 0;                                    // stride hi | dim1_stride lo
    g1[7] = 0;
    i32x4 gz4 = {0, 0, 0, 0};
    i32x8 gz8 = {0, 0, 0, 0, 0, 0, 0, 0};
    // 6-arg toolchain variant: (g0, g1, g2, g3, extra, cpol)
    __builtin_amdgcn_tensor_load_to_lds(g0, g1, gz4, gz4, gz8, 0);
}

// ---------------------------------------------------------------------------
// Prep kernels
// ---------------------------------------------------------------------------

// X (N x D, f32 row-major) -> Xt (D x N, bf16)  [B-operand friendly layout]
__global__ void k_xt(const float* __restrict__ X, __bf16* __restrict__ Xt) {
    int idx = blockIdx.x * 256 + threadIdx.x;   // idx = j*64 + d
    int j = idx >> 6, d = idx & 63;
    Xt[d * N + j] = (__bf16)X[idx];
}

// gc1w / gc2w (64x64 f32 [out,in]) -> bf16 row-major (B[k][n] = W[n][k] friendly)
__global__ void k_wbf(const float* __restrict__ w1, const float* __restrict__ w2,
                      __bf16* __restrict__ o1, __bf16* __restrict__ o2) {
    int i = blockIdx.x * 256 + threadIdx.x;
    if (i < 64 * 64) { o1[i] = (__bf16)w1[i]; o2[i] = (__bf16)w2[i]; }
}

// v1[l][k] = sum_d Ww[l][d][k] * attw[l][d] ;  v2 with attw[l][D+d]
__global__ void k_vvec(const float* __restrict__ Ww, const float* __restrict__ Wb,
                       const float* __restrict__ attw,
                       float* __restrict__ v1, float* __restrict__ v2,
                       float* __restrict__ c12) {
    int l = blockIdx.x, k = threadIdx.x;       // grid=3, block=64
    float a1 = 0.f, a2 = 0.f;
    for (int d = 0; d < D; ++d) {
        float w = Ww[(l * D + d) * D + k];
        a1 += w * attw[l * 2 * D + d];
        a2 += w * attw[l * 2 * D + D + d];
    }
    v1[l * D + k] = a1;
    v2[l * D + k] = a2;
    if (k == 0) {
        float b1 = 0.f, b2 = 0.f;
        for (int d = 0; d < D; ++d) {
            b1 += Wb[l * D + d] * attw[l * 2 * D + d];
            b2 += Wb[l * D + d] * attw[l * 2 * D + D + d];
        }
        c12[l * 2] = b1;
        c12[l * 2 + 1] = b2;
    }
}

// s1[l][i] = X[i,:].v1[l] + c1[l] ; s2 likewise
__global__ void k_scores(const float* __restrict__ X,
                         const float* __restrict__ v1, const float* __restrict__ v2,
                         const float* __restrict__ c12,
                         float* __restrict__ s1, float* __restrict__ s2) {
    int i = blockIdx.x * blockDim.x + threadIdx.x;
    if (i >= N) return;
    const float* x = X + i * D;
    #pragma unroll
    for (int l = 0; l < L3; ++l) {
        float a = c12[2 * l], b = c12[2 * l + 1];
        for (int k = 0; k < D; ++k) {
            float xv = x[k];
            a += xv * v1[l * D + k];
            b += xv * v2[l * D + k];
        }
        s1[l * N + i] = a;
        s2[l * N + i] = b;
    }
}

// maxs2[l] = max_j s2[l][j]  (grid=3, block=256)
__global__ void k_maxred(const float* __restrict__ s2, float* __restrict__ maxs2) {
    __shared__ float red[256];
    int l = blockIdx.x;
    float m = -INFINITY;
    for (int i = threadIdx.x; i < N; i += 256) m = fmaxf(m, s2[l * N + i]);
    red[threadIdx.x] = m;
    __syncthreads();
    for (int s = 128; s > 0; s >>= 1) {
        if (threadIdx.x < s) red[threadIdx.x] = fmaxf(red[threadIdx.x], red[threadIdx.x + s]);
        __syncthreads();
    }
    if (threadIdx.x == 0) maxs2[l] = red[0];
}

// ---------------------------------------------------------------------------
// MAS: X_MAS = A_MAS @ X, flash-attention style, WMMA bf16 + TDM-staged B.
// One wave owns a 16-row tile; A fragment = exp weights built in-register.
// grid = N/32 blocks of 64 threads (2 waves).
// ---------------------------------------------------------------------------
__global__ void __launch_bounds__(64) k_mas(const float* __restrict__ s1,
                                            const float* __restrict__ s2,
                                            const float* __restrict__ maxs2,
                                            const float* __restrict__ attb,
                                            const __bf16* __restrict__ Xt,
                                            __bf16* __restrict__ XmasT) {
    __shared__ float zbuf[2][16];
    __shared__ __align__(32) __bf16 bpanel[2][2][D * 32];   // [wave][buf][d*32+jj]
    const int wave = threadIdx.x >> 5, lane = threadIdx.x & 31;
    const int m = lane & 15, h = lane >> 4;
    const int rowbase = blockIdx.x * 32 + wave * 16;
    const int row = rowbase + m;

    float attbv[L3], s1v[L3], Mrow = 0.f;
    #pragma unroll
    for (int l = 0; l < L3; ++l) {
        attbv[l] = attb[l];
        s1v[l]   = s1[l * N + row];
        Mrow    += lrelu(s1v[l] + maxs2[l] + attbv[l]);   // fixed row max (monotone lrelu)
    }

    const unsigned int loff[2] = { lds_offset(&bpanel[wave][0][0]),
                                   lds_offset(&bpanel[wave][1][0]) };
    tdm_load_panel(Xt, loff[0]);                          // prime buffer 0

    v8f acc[4] = {};
    float zpart = 0.f;

    for (int jb = 0; jb < N; jb += 32) {
        const int buf = (jb >> 5) & 1;
        if (jb + 32 < N) {                                // prefetch next via TDM
            tdm_load_panel(Xt + jb + 32, loff[buf ^ 1]);
            __builtin_amdgcn_s_wait_tensorcnt(1);         // current buf ready
        } else {
            __builtin_amdgcn_s_wait_tensorcnt(0);
        }

        FragBF a;
        float psum = 0.f;
        #pragma unroll
        for (int i = 0; i < 16; ++i) {
            const int col = jb + kmapA(i, h);
            float t = 0.f;
            #pragma unroll
            for (int l = 0; l < L3; ++l)
                t += lrelu(s1v[l] + s2[l * N + col] + attbv[l]);
            float p = __expf(t - Mrow);     // single exp per (i,j)
            psum += p;
            a.e[i] = (__bf16)p;
        }
        zpart += psum;
        #pragma unroll
        for (int nt = 0; nt < 4; ++nt) {
            const bf16x16* bp =
                (const bf16x16*)&bpanel[wave][buf][(nt * 16 + m) * 32 + h * 16];
            acc[nt] = __builtin_amdgcn_wmma_f32_16x16x32_bf16(
                false, a.v, false, *bp, (short)0, acc[nt], false, false);
        }
    }

    // combine the two K-halves of the row sums, share across the wave via LDS
    zpart += __shfl_xor(zpart, 16, 32);
    if (lane < 16) zbuf[wave][lane] = zpart;
    __syncthreads();

    #pragma unroll
    for (int nt = 0; nt < 4; ++nt) {
        Pack8 pk;
        #pragma unroll
        for (int r = 0; r < 8; ++r) {
            float vl = acc[nt][r] / zbuf[wave][r + 8 * h];
            pk.e[r] = (__bf16)vl;
        }
        // store transposed bf16: XmasT[d][row], d = nt*16 + m (lane's D column)
        *(uint4*)(XmasT + (size_t)(nt * 16 + m) * N + rowbase + 8 * h) = pk.q;
    }
}

// ---------------------------------------------------------------------------
// GNFE big GEMM: Out = relu((A_hat @ V) @ W^T + bias)
//   A read as f32 (HBM stream), +I on diagonal, converted to bf16 in-reg.
//   V supplied transposed bf16 (Bt), staged through LDS by the TDM.
//   Epilogue small GEMM done with WMMA after an LDS C->A relayout.
// grid = N/32 blocks of 64 threads.
// ---------------------------------------------------------------------------
__global__ void __launch_bounds__(64) k_spmm(const float* __restrict__ A,
                                             const __bf16* __restrict__ Bt,
                                             const __bf16* __restrict__ Wbf,
                                             const float* __restrict__ bias,
                                             __bf16* __restrict__ outT,
                                             float* __restrict__ outF) {
    __shared__ float tile[2][16][68];           // +4 pad: conflict-free column reads
    __shared__ __align__(32) __bf16 bpanel[2][2][D * 32];
    const int wave = threadIdx.x >> 5, lane = threadIdx.x & 31;
    const int m = lane & 15, h = lane >> 4;
    const int rowbase = blockIdx.x * 32 + wave * 16;
    const int row = rowbase + m;
    const float* Arow = A + (size_t)row * N;

    const unsigned int loff[2] = { lds_offset(&bpanel[wave][0][0]),
                                   lds_offset(&bpanel[wave][1][0]) };
    tdm_load_panel(Bt, loff[0]);

    v8f acc[4] = {};
    for (int jb = 0; jb < N; jb += 32) {
        const int buf = (jb >> 5) & 1;
        if (jb + 32 < N) {
            tdm_load_panel(Bt + jb + 32, loff[buf ^ 1]);
            __builtin_amdgcn_s_wait_tensorcnt(1);
        } else {
            __builtin_amdgcn_s_wait_tensorcnt(0);
        }
        if (jb + 128 < N) __builtin_prefetch(Arow + jb + 128, 0, 0);

        FragBF a;
        #pragma unroll
        for (int t = 0; t < 8; ++t) {           // 8 contiguous K-pairs -> float2 loads
            int k0 = 2 * t + 16 * h;
            float2 f = *(const float2*)(Arow + jb + k0);
            int c0 = jb + k0;
            if (c0 == row)     f.x += 1.0f;     // A_hat = A + I
            if (c0 + 1 == row) f.y += 1.0f;
            a.e[2 * t]     = (__bf16)f.x;
            a.e[2 * t + 1] = (__bf16)f.y;
        }
        #pragma unroll
        for (int nt = 0; nt < 4; ++nt) {
            const bf16x16* bp =
                (const bf16x16*)&bpanel[wave][buf][(nt * 16 + m) * 32 + h * 16];
            acc[nt] = __builtin_amdgcn_wmma_f32_16x16x32_bf16(
                false, a.v, false, *bp, (short)0, acc[nt], false, false);
        }
    }

    // C/D layout -> LDS (row r+8h, col nt*16+n), then rebuild as A fragments
    #pragma unroll
    for (int nt = 0; nt < 4; ++nt)
        #pragma unroll
        for (int r = 0; r < 8; ++r)
            tile[wave][r + 8 * h][nt * 16 + m] = acc[nt][r];
    __syncthreads();

    FragBF af[2];
    #pragma unroll
    for (int c = 0; c < 2; ++c)
        #pragma unroll
        for (int i = 0; i < 16; ++i)
            af[c].e[i] = (__bf16)tile[wave][m][c * 32 + kmapA(i, h)];

    v8f acc2[4] = {};
    #pragma unroll
    for (int nt = 0; nt < 4; ++nt) {
        #pragma unroll
        for (int c = 0; c < 2; ++c) {
            const bf16x16* wp = (const bf16x16*)(Wbf + (nt * 16 + m) * 64 + c * 32 + h * 16);
            acc2[nt] = __builtin_amdgcn_wmma_f32_16x16x32_bf16(
                false, af[c].v, false, *wp, (short)0, acc2[nt], false, false);
        }
    }

    #pragma unroll
    for (int nt = 0; nt < 4; ++nt) {
        float bv = bias[nt * 16 + m];
        if (outT) {
            Pack8 pk;
            #pragma unroll
            for (int r = 0; r < 8; ++r)
                pk.e[r] = (__bf16)fmaxf(acc2[nt][r] + bv, 0.f);
            *(uint4*)(outT + (size_t)(nt * 16 + m) * N + rowbase + 8 * h) = pk.q;
        }
        if (outF) {
            #pragma unroll
            for (int r = 0; r < 8; ++r)
                outF[(size_t)(rowbase + r + 8 * h) * D + nt * 16 + m] =
                    fmaxf(acc2[nt][r] + bv, 0.f);
        }
    }
}

// ---------------------------------------------------------------------------
// Final: hard gumbel-softmax feature selection. grid = N, block = 64 (one row)
// ---------------------------------------------------------------------------
__global__ void __launch_bounds__(64) k_final(const float* __restrict__ H,
                                              const float* __restrict__ gsw,
                                              const float* __restrict__ gsb,
                                              const float* __restrict__ gum,
                                              float* __restrict__ out) {
    __shared__ float hrow[64];
    __shared__ float red[64];
    __shared__ int   ridx[64];
    const int r = blockIdx.x, n = threadIdx.x;
    hrow[n] = H[r * D + n];
    __syncthreads();

    float logit = gsb[n];
    for (int k = 0; k < D; ++k) logit += hrow[k] * gsw[n * D + k];
    float g = (logit + gum[r * D + n]) * 2.0f;            // / TAU (0.5)

    red[n] = g; __syncthreads();
    for (int s = 32; s > 0; s >>= 1) {
        if (n < s) red[n] = fmaxf(red[n], red[n + s]);
        __syncthreads();
    }
    float mx = red[0]; __syncthreads();

    float ex = __expf(g - mx);
    red[n] = ex; __syncthreads();
    for (int s = 32; s > 0; s >>= 1) {
        if (n < s) red[n] += red[n + s];
        __syncthreads();
    }
    float Zs = red[0]; __syncthreads();
    float ys = ex / Zs;

    // argmax(y_soft) == argmax(g) (softmax is monotone); first-occurrence ties
    red[n] = g; ridx[n] = n; __syncthreads();
    for (int s = 32; s > 0; s >>= 1) {
        if (n < s) {
            float ov = red[n + s]; int oi = ridx[n + s];
            if (ov > red[n] || (ov == red[n] && oi < ridx[n])) { red[n] = ov; ridx[n] = oi; }
        }
        __syncthreads();
    }
    int am = ridx[0];
    float yh = (n == am) ? 1.0f : 0.0f;
    float sel = (yh - ys) + ys;                           // straight-through value
    out[r * D + n] = hrow[n] * sel;
}

// ---------------------------------------------------------------------------
// Launch
// ---------------------------------------------------------------------------
extern "C" void kernel_launch(void* const* d_in, const int* in_sizes, int n_in,
                              void* d_out, int out_size, void* d_ws, size_t ws_size,
                              hipStream_t stream) {
    const float* X     = (const float*)d_in[0];
    const float* A     = (const float*)d_in[1];
    const float* Ww    = (const float*)d_in[2];
    const float* Wb    = (const float*)d_in[3];
    const float* attw  = (const float*)d_in[4];
    const float* attb  = (const float*)d_in[5];
    const float* gc1w  = (const float*)d_in[6];
    const float* gc1b  = (const float*)d_in[7];
    const float* gc2w  = (const float*)d_in[8];
    const float* gc2b  = (const float*)d_in[9];
    const float* gsw   = (const float*)d_in[10];
    const float* gsb   = (const float*)d_in[11];
    const float* gum   = (const float*)d_in[12];
    float* out = (float*)d_out;

    char* base = (char*)d_ws;
    auto alloc = [&](size_t bytes) -> void* {
        void* p = base;
        base += (bytes + 255) & ~(size_t)255;
        return p;
    };
    __bf16* Xt    = (__bf16*)alloc((size_t)D * N * sizeof(__bf16));
    __bf16* XmasT = (__bf16*)alloc((size_t)D * N * sizeof(__bf16));
    __bf16* H1T   = (__bf16*)alloc((size_t)D * N * sizeof(__bf16));
    float*  Hf    = (float*) alloc((size_t)N * D * sizeof(float));
    float*  s1    = (float*) alloc((size_t)L3 * N * sizeof(float));
    float*  s2    = (float*) alloc((size_t)L3 * N * sizeof(float));
    float*  v1    = (float*) alloc((size_t)L3 * D * sizeof(float));
    float*  v2    = (float*) alloc((size_t)L3 * D * sizeof(float));
    float*  c12   = (float*) alloc(2 * L3 * sizeof(float));
    float*  mxs2  = (float*) alloc(L3 * sizeof(float));
    __bf16* Wbf1  = (__bf16*)alloc((size_t)D * D * sizeof(__bf16));
    __bf16* Wbf2  = (__bf16*)alloc((size_t)D * D * sizeof(__bf16));

    k_xt     <<<(N * D) / 256, 256, 0, stream>>>(X, Xt);
    k_wbf    <<<(D * D + 255) / 256, 256, 0, stream>>>(gc1w, gc2w, Wbf1, Wbf2);
    k_vvec   <<<L3, D, 0, stream>>>(Ww, Wb, attw, v1, v2, c12);
    k_scores <<<N / 256, 256, 0, stream>>>(X, v1, v2, c12, s1, s2);
    k_maxred <<<L3, 256, 0, stream>>>(s2, mxs2);

    k_mas    <<<N / 32, 64, 0, stream>>>(s1, s2, mxs2, attb, Xt, XmasT);

    // H1 = relu((A_hat @ X_MAS) @ gc1w^T + gc1b)  -> bf16 transposed
    k_spmm   <<<N / 32, 64, 0, stream>>>(A, XmasT, Wbf1, gc1b, H1T, nullptr);
    // H  = relu((A_hat @ H1) @ gc2w^T + gc2b)     -> f32 row-major
    k_spmm   <<<N / 32, 64, 0, stream>>>(A, H1T, Wbf2, gc2b, nullptr, Hf);

    k_final  <<<N, 64, 0, stream>>>(Hf, gsw, gsb, gum, out);
}
